// DecoderLayer_29695403885041
// MI455X (gfx1250) — matrix operand order repaired and checked by reading, not compiled, not run on hardware
//
#include <hip/hip_runtime.h>
#include <math.h>
#include <stdint.h>

typedef __attribute__((ext_vector_type(16))) _Float16 v16h;
typedef __attribute__((ext_vector_type(8)))  _Float16 v8h;
typedef __attribute__((ext_vector_type(8)))  float    v8f;

#define D_MODEL 1024
#define N_HEADS 16
#define DK      64
#define D_FF    4096
#define B_      4
#define L_      1024
#define NTOK    4096     // B_ * L_
#define EPS     1e-5f

// ---------------------------------------------------------------------------
// WMMA + fragment helpers (gfx1250 wave32)
// ---------------------------------------------------------------------------
static __device__ __forceinline__ v8f wmma_f16(v16h a, v16h b, v8f c) {
  return __builtin_amdgcn_wmma_f32_16x16x32_f16(false, a, false, b, (short)0, c,
                                                false, false);
}

// A fragment from LDS, row-major 16x32 tile (stride in f16 elems).
static __device__ __forceinline__ v16h frag_a_lds(const _Float16* base,
                                                  int stride, int lane) {
  int row = lane & 15;
  int kb  = (lane & 16) ? 8 : 0;
  const _Float16* p = base + row * stride + kb;
  v8h lo = *reinterpret_cast<const v8h*>(p);
  v8h hi = *reinterpret_cast<const v8h*>(p + 16);
  v16h r;
#pragma unroll
  for (int i = 0; i < 8; ++i) { r[i] = lo[i]; r[i + 8] = hi[i]; }
  return r;
}

// A fragment directly from global f16 (Q tiles), scaled by 1/sqrt(dk)=0.125.
static __device__ __forceinline__ v16h frag_a_gmem_h(const _Float16* base,
                                                     int stride, int lane) {
  int row = lane & 15;
  int kb  = (lane & 16) ? 8 : 0;
  const _Float16* p = base + (size_t)row * stride + kb;
  v8h lo = *reinterpret_cast<const v8h*>(p);
  v8h hi = *reinterpret_cast<const v8h*>(p + 16);
  v16h r;
#pragma unroll
  for (int i = 0; i < 8; ++i) {
    r[i]     = lo[i] * (_Float16)0.125f;
    r[i + 8] = hi[i] * (_Float16)0.125f;
  }
  return r;
}

// B fragment from LDS stored TRANSPOSED as [n][k] (stride in f16 elems).
static __device__ __forceinline__ v16h frag_b_ldsT(const _Float16* base,
                                                   int stride, int lane) {
  int col = lane & 15;
  int kb  = (lane & 16) ? 8 : 0;
  const _Float16* p = base + col * stride + kb;
  v8h lo = *reinterpret_cast<const v8h*>(p);
  v8h hi = *reinterpret_cast<const v8h*>(p + 16);
  v16h r;
#pragma unroll
  for (int i = 0; i < 8; ++i) { r[i] = lo[i]; r[i + 8] = hi[i]; }
  return r;
}

// async 16B global->LDS copy (ASYNCcnt tracked)
static __device__ __forceinline__ void async_cp16(const _Float16* g,
                                                  _Float16* lds) {
  unsigned l = (unsigned)(uintptr_t)lds;  // low 32 bits = LDS byte offset
  asm volatile("global_load_async_to_lds_b128 %0, %1, off"
               :: "v"(l), "v"(g) : "memory");
}

// ---------------------------------------------------------------------------
// f32 -> f16 elementwise convert
// ---------------------------------------------------------------------------
__global__ __launch_bounds__(256)
void cvt_f16_kernel(const float* __restrict__ in, _Float16* __restrict__ out,
                    int n) {
  int i = (blockIdx.x * 256 + threadIdx.x) * 4;
#pragma unroll
  for (int j = 0; j < 4; ++j) out[i + j] = (_Float16)in[i + j];
}

// ---------------------------------------------------------------------------
// W[K,N] f32 -> Wt[N,K] f16 (tiled transpose through LDS)
// ---------------------------------------------------------------------------
__global__ __launch_bounds__(256)
void transpose_cvt_kernel(const float* __restrict__ in,
                          _Float16* __restrict__ out, int K, int N) {
  __shared__ _Float16 tt[64 * 72];
  const int tid = threadIdx.x;
  const int n0 = blockIdx.x * 64, k0 = blockIdx.y * 64;
#pragma unroll
  for (int t = 0; t < 16; ++t) {
    int idx = tid + t * 256;
    int kk = idx >> 6, nn = idx & 63;        // coalesced along n
    tt[nn * 72 + kk] = (_Float16)in[(size_t)(k0 + kk) * N + n0 + nn];
  }
  __syncthreads();
#pragma unroll
  for (int t = 0; t < 16; ++t) {
    int idx = tid + t * 256;
    int nn = idx >> 6, kk = idx & 63;        // coalesced along k
    out[(size_t)(n0 + nn) * K + k0 + kk] = tt[nn * 72 + kk];
  }
}

// ---------------------------------------------------------------------------
// GEMM: out = act(A[M,K](f16) @ Wt[N,K](f16)^T + bias)
// 256 thr / 8 waves; block tile 128x128; wave tile 32x64; k-step 64.
// Double-buffered LDS filled with global_load_async_to_lds_b128.
// ---------------------------------------------------------------------------
__global__ __launch_bounds__(256)
void gemm_f16_kernel(const _Float16* __restrict__ A,
                     const _Float16* __restrict__ Wt,
                     const float* __restrict__ bias,
                     float* __restrict__ out32, _Float16* __restrict__ out16,
                     int M, int N, int K, int relu) {
  __shared__ __align__(16) _Float16 As[2][128 * 72];  // [row][k] stride 72
  __shared__ __align__(16) _Float16 Bs[2][128 * 72];  // [n][k]  stride 72
  const int tid  = threadIdx.x;
  const int lane = tid & 31;
  const int wave = tid >> 5;
  const int row0 = blockIdx.y * 128;
  const int col0 = blockIdx.x * 128;
  const int wr = (wave & 3) * 32;
  const int wc = (wave >> 2) * 64;
  const int T = K >> 6;

  auto issue = [&](int it) {
    const int k0 = it << 6;
    _Float16* ab = As[it & 1];
    _Float16* bb = Bs[it & 1];
#pragma unroll
    for (int t = 0; t < 4; ++t) {            // A: 128 rows x 8 chunks = 1024
      int c = tid + t * 256;
      int r = c >> 3, cc = (c & 7) * 8;
      async_cp16(A + (size_t)(row0 + r) * K + k0 + cc, ab + r * 72 + cc);
    }
#pragma unroll
    for (int t = 0; t < 4; ++t) {            // B: 128 rows x 8 chunks = 1024
      int c = tid + t * 256;
      int r = c >> 3, cc = (c & 7) * 8;
      async_cp16(Wt + (size_t)(col0 + r) * K + k0 + cc, bb + r * 72 + cc);
    }
  };

  v8f zero = {};
  v8f acc[2][4];
#pragma unroll
  for (int ii = 0; ii < 2; ++ii)
#pragma unroll
    for (int jj = 0; jj < 4; ++jj) acc[ii][jj] = zero;

  issue(0);
  for (int i = 0; i < T; ++i) {
    if (i + 1 < T) {
      issue(i + 1);
      asm volatile("s_wait_asynccnt 0x8" ::: "memory");  // tile i landed
    } else {
      asm volatile("s_wait_asynccnt 0x0" ::: "memory");
    }
    __syncthreads();
    const _Float16* ab = As[i & 1];
    const _Float16* bb = Bs[i & 1];
#pragma unroll
    for (int ks = 0; ks < 64; ks += 32) {
      v16h a0 = frag_a_lds(ab + wr * 72 + ks, 72, lane);
      v16h a1 = frag_a_lds(ab + (wr + 16) * 72 + ks, 72, lane);
#pragma unroll
      for (int jj = 0; jj < 4; ++jj) {
        v16h b = frag_b_ldsT(bb + (wc + jj * 16) * 72 + ks, 72, lane);
        acc[0][jj] = wmma_f16(a0, b, acc[0][jj]);
        acc[1][jj] = wmma_f16(a1, b, acc[1][jj]);
      }
    }
    __syncthreads();
  }

  const int rb = (lane & 16) ? 8 : 0;
  const int cb = lane & 15;
#pragma unroll
  for (int ii = 0; ii < 2; ++ii)
#pragma unroll
    for (int jj = 0; jj < 4; ++jj) {
#pragma unroll
      for (int i = 0; i < 8; ++i) {
        int r = row0 + wr + ii * 16 + rb + i;
        int c = col0 + wc + jj * 16 + cb;
        float v = acc[ii][jj][i] + bias[c];
        if (relu) v = fmaxf(v, 0.f);
        if (out32) out32[(size_t)r * N + c] = v;
        if (out16) out16[(size_t)r * N + c] = (_Float16)v;
      }
    }
}

// ---------------------------------------------------------------------------
// T5 relative-position bias by distance d = q - k >= 0
// ---------------------------------------------------------------------------
static __device__ __forceinline__ float t5_bias_d(int d,
                                                  const float* __restrict__ rel,
                                                  int h) {
  int bucket;
  if (d < 16) {
    bucket = d;
  } else {
    float vl = 16.0f + logf((float)d * 0.0625f) * (16.0f / 2.0794415416798357f);
    int bi = (int)vl;
    bucket = bi < 31 ? bi : 31;
  }
  return rel[bucket * N_HEADS + h];
}

// ---------------------------------------------------------------------------
// Flash attention (f16 Q/K/V/ctx): 1 wave / 16 q-rows, 8 waves / block.
// Double-buffered K/V staging; K via async b128; per-head bias LDS table.
// ---------------------------------------------------------------------------
template <bool CAUSAL>
__global__ __launch_bounds__(256)
void attn_kernel(const _Float16* __restrict__ Qb, const _Float16* __restrict__ Kb,
                 const _Float16* __restrict__ Vb, _Float16* __restrict__ Cb,
                 const float* __restrict__ rel_emb, const int* __restrict__ kmask,
                 int Lq, int Lk) {
  __shared__ __align__(16) _Float16 Ks[2][64 * 72];   // [kpos][dk]
  __shared__ __align__(16) _Float16 Vs[2][64 * 72];   // [dv][kpos]
  __shared__ __align__(16) _Float16 Ps[8 * 16 * 40];  // per-wave P tile 16x32
  __shared__ float bias_tab[1024];                    // distance -> bias (CAUSAL)

  const int tid = threadIdx.x, lane = tid & 31, wave = tid >> 5;
  const int bb = blockIdx.y / N_HEADS;
  const int h  = blockIdx.y % N_HEADS;
  const int qb0 = blockIdx.x * 128;
  const int q0  = qb0 + wave * 16;
  const int qmax = q0 + 15;
  _Float16* Pw = Ps + wave * 16 * 40;

  if (CAUSAL) {
#pragma unroll
    for (int t = 0; t < 4; ++t) {
      int d = tid + t * 256;
      bias_tab[d] = t5_bias_d(d, rel_emb, h);
    }
  }

  const _Float16* qptr = Qb + ((size_t)(bb * Lq + q0)) * D_MODEL + h * DK;
  v16h qf0 = frag_a_gmem_h(qptr, D_MODEL, lane);
  v16h qf1 = frag_a_gmem_h(qptr + 32, D_MODEL, lane);

  v8f o0 = {}, o1 = {}, o2 = {}, o3 = {};
  float mrow[8], lrow[8];
#pragma unroll
  for (int i = 0; i < 8; ++i) { mrow[i] = -3.0e38f; lrow[i] = 0.0f; }

  const int rb = (lane & 16) ? 8 : 0;
  const int cb = lane & 15;
  const int klimit = CAUSAL ? (qb0 + 128 < Lk ? qb0 + 128 : Lk) : Lk;
  const int nch = klimit >> 6;

  auto issueKV = [&](int it) {
    const int kt = it << 6;
    _Float16* ks = Ks[it & 1];
    _Float16* vs = Vs[it & 1];
#pragma unroll
    for (int t = 0; t < 2; ++t) {            // K: async, 512 x 16B chunks
      int c = tid + t * 256;
      int kp = c >> 3, cc = (c & 7) * 8;
      async_cp16(Kb + ((size_t)(bb * Lk + kt + kp)) * D_MODEL + h * DK + cc,
                 ks + kp * 72 + cc);
    }
#pragma unroll
    for (int t = 0; t < 2; ++t) {            // V: manual transpose into LDS
      int c = tid + t * 256;
      int kp = c >> 3, dc = (c & 7) * 8;
      v8h val = *reinterpret_cast<const v8h*>(
          Vb + ((size_t)(bb * Lk + kt + kp)) * D_MODEL + h * DK + dc);
#pragma unroll
      for (int j = 0; j < 8; ++j) vs[(dc + j) * 72 + kp] = val[j];
    }
  };

  issueKV(0);
  for (int it = 0; it < nch; ++it) {
    const int kt = it << 6;
    if (it + 1 < nch) {
      issueKV(it + 1);
      asm volatile("s_wait_asynccnt 0x2" ::: "memory");  // chunk it landed
    } else {
      asm volatile("s_wait_asynccnt 0x0" ::: "memory");
    }
    __syncthreads();
    if (!(CAUSAL && kt > qmax)) {
      const _Float16* ks = Ks[it & 1];
      const _Float16* vs = Vs[it & 1];
#pragma unroll
      for (int kc = 0; kc < 64; kc += 32) {
        if (CAUSAL && kt + kc > qmax) break;

        v8f s0 = {}, s1 = {};
        s0 = wmma_f16(qf0, frag_b_ldsT(ks + (kc + 0) * 72 + 0, 72, lane), s0);
        s0 = wmma_f16(qf1, frag_b_ldsT(ks + (kc + 0) * 72 + 32, 72, lane), s0);
        s1 = wmma_f16(qf0, frag_b_ldsT(ks + (kc + 16) * 72 + 0, 72, lane), s1);
        s1 = wmma_f16(qf1, frag_b_ldsT(ks + (kc + 16) * 72 + 32, 72, lane), s1);

        const int k0c = kt + kc + cb;
        const int k1c = k0c + 16;
        bool m0 = true, m1 = true;
        if (!CAUSAL) {
          m0 = kmask[(size_t)bb * Lk + k0c] != 0;
          m1 = kmask[(size_t)bb * Lk + k1c] != 0;
        }
#pragma unroll
        for (int i = 0; i < 8; ++i) {
          int q = q0 + rb + i;
          float v0 = s0[i], v1 = s1[i];
          if (CAUSAL) {
            v0 = (k0c <= q) ? v0 + bias_tab[(q - k0c) & 1023] : -1e9f;
            v1 = (k1c <= q) ? v1 + bias_tab[(q - k1c) & 1023] : -1e9f;
          } else {
            v0 = m0 ? v0 : -1e9f;
            v1 = m1 ? v1 : -1e9f;
          }
          float cm = fmaxf(v0, v1);
          cm = fmaxf(cm, __shfl_xor(cm, 1, 32));
          cm = fmaxf(cm, __shfl_xor(cm, 2, 32));
          cm = fmaxf(cm, __shfl_xor(cm, 4, 32));
          cm = fmaxf(cm, __shfl_xor(cm, 8, 32));
          float mn = fmaxf(mrow[i], cm);
          float corr = __expf(mrow[i] - mn);
          mrow[i] = mn;
          float p0 = __expf(v0 - mn);
          float p1 = __expf(v1 - mn);
          float ps = p0 + p1;
          ps += __shfl_xor(ps, 1, 32);
          ps += __shfl_xor(ps, 2, 32);
          ps += __shfl_xor(ps, 4, 32);
          ps += __shfl_xor(ps, 8, 32);
          lrow[i] = lrow[i] * corr + ps;
          o0[i] *= corr; o1[i] *= corr; o2[i] *= corr; o3[i] *= corr;
          Pw[(rb + i) * 40 + cb]      = (_Float16)p0;
          Pw[(rb + i) * 40 + 16 + cb] = (_Float16)p1;
        }
        asm volatile("s_wait_dscnt 0x0" ::: "memory");
        v16h pf = frag_a_lds(Pw, 40, lane);
        o0 = wmma_f16(pf, frag_b_ldsT(vs + 0 * 16 * 72 + kc, 72, lane), o0);
        o1 = wmma_f16(pf, frag_b_ldsT(vs + 1 * 16 * 72 + kc, 72, lane), o1);
        o2 = wmma_f16(pf, frag_b_ldsT(vs + 2 * 16 * 72 + kc, 72, lane), o2);
        o3 = wmma_f16(pf, frag_b_ldsT(vs + 3 * 16 * 72 + kc, 72, lane), o3);
      }
    }
    __syncthreads();
  }

#pragma unroll
  for (int i = 0; i < 8; ++i) {
    float inv = lrow[i] > 0.0f ? 1.0f / lrow[i] : 0.0f;
    size_t base = ((size_t)(bb * Lq + q0 + rb + i)) * D_MODEL + h * DK + cb;
    Cb[base + 0]  = (_Float16)(o0[i] * inv);
    Cb[base + 16] = (_Float16)(o1[i] * inv);
    Cb[base + 32] = (_Float16)(o2[i] * inv);
    Cb[base + 48] = (_Float16)(o3[i] * inv);
  }
}

// ---------------------------------------------------------------------------
// outf32/outf16 = LayerNorm(x + r) * g + b  (one block per row, D=1024)
// ---------------------------------------------------------------------------
__global__ __launch_bounds__(256)
void add_ln_kernel(const float* __restrict__ x, const float* __restrict__ r,
                   const float* __restrict__ g, const float* __restrict__ be,
                   float* __restrict__ outf, _Float16* __restrict__ outh) {
  __shared__ float red[256];
  const int row = blockIdx.x, tid = threadIdx.x;
  const size_t base = (size_t)row * D_MODEL;
  float v[4];
  float s = 0.f;
#pragma unroll
  for (int j = 0; j < 4; ++j) {
    int c = tid + j * 256;
    float t = x[base + c] + r[base + c];
    v[j] = t; s += t;
  }
  red[tid] = s; __syncthreads();
  for (int o = 128; o > 0; o >>= 1) {
    if (tid < o) red[tid] += red[tid + o];
    __syncthreads();
  }
  float mu = red[0] * (1.0f / D_MODEL);
  __syncthreads();
  float sq = 0.f;
#pragma unroll
  for (int j = 0; j < 4; ++j) { float d = v[j] - mu; sq += d * d; }
  red[tid] = sq; __syncthreads();
  for (int o = 128; o > 0; o >>= 1) {
    if (tid < o) red[tid] += red[tid + o];
    __syncthreads();
  }
  float rs = rsqrtf(red[0] * (1.0f / D_MODEL) + EPS);
#pragma unroll
  for (int j = 0; j < 4; ++j) {
    int c = tid + j * 256;
    float val = (v[j] - mu) * rs * g[c] + be[c];
    outf[base + c] = val;
    if (outh) outh[base + c] = (_Float16)val;
  }
}

// ---------------------------------------------------------------------------
// Host-side orchestration
// ---------------------------------------------------------------------------
extern "C" void kernel_launch(void* const* d_in, const int* in_sizes, int n_in,
                              void* d_out, int out_size, void* d_ws, size_t ws_size,
                              hipStream_t stream) {
  const float* x        = (const float*)d_in[0];
  const float* mem      = (const float*)d_in[1];
  const int*   mem_mask = (const int*)d_in[3];
  const float* sa_wq = (const float*)d_in[4];
  const float* sa_bq = (const float*)d_in[5];
  const float* sa_wk = (const float*)d_in[6];
  const float* sa_bk = (const float*)d_in[7];
  const float* sa_wv = (const float*)d_in[8];
  const float* sa_bv = (const float*)d_in[9];
  const float* sa_wo = (const float*)d_in[10];
  const float* sa_bo = (const float*)d_in[11];
  const float* rel_emb = (const float*)d_in[12];
  const float* ca_wq = (const float*)d_in[13];
  const float* ca_bq = (const float*)d_in[14];
  const float* ca_wk = (const float*)d_in[15];
  const float* ca_bk = (const float*)d_in[16];
  const float* ca_wv = (const float*)d_in[17];
  const float* ca_bv = (const float*)d_in[18];
  const float* ca_wo = (const float*)d_in[19];
  const float* ca_bo = (const float*)d_in[20];
  const float* fc1_w = (const float*)d_in[21];
  const float* fc1_b = (const float*)d_in[22];
  const float* fc2_w = (const float*)d_in[23];
  const float* fc2_b = (const float*)d_in[24];
  const float* ln1_g = (const float*)d_in[25];
  const float* ln1_b = (const float*)d_in[26];
  const float* ln2_g = (const float*)d_in[27];
  const float* ln2_b = (const float*)d_in[28];
  const float* ln3_g = (const float*)d_in[29];
  const float* ln3_b = (const float*)d_in[30];

  // ---- workspace arena (byte offsets) ----
  char* wsb = (char*)d_ws;
  const size_t MB = 1u << 20;
  _Float16* Qh   = (_Float16*)(wsb + 0);        // 8MB
  _Float16* Kh   = (_Float16*)(wsb + 8 * MB);   // 8MB
  _Float16* Vh   = (_Float16*)(wsb + 16 * MB);  // 8MB
  _Float16* Ch   = (_Float16*)(wsb + 24 * MB);  // 8MB
  _Float16* FFh  = (_Float16*)(wsb + 0);        // 32MB overlay (FFN phase only)
  float*    TMP  = (float*)(wsb + 32 * MB);     // 16MB
  float*    X1f  = (float*)(wsb + 48 * MB);     // 16MB
  float*    X2f  = (float*)(wsb + 64 * MB);     // 16MB
  _Float16* xh   = (_Float16*)(wsb + 80 * MB);  // 8MB
  _Float16* memh = (_Float16*)(wsb + 88 * MB);  // 8MB
  _Float16* X1h  = (_Float16*)(wsb + 96 * MB);  // 8MB
  _Float16* X2h  = (_Float16*)(wsb + 104 * MB); // 8MB
  _Float16* Wsaq = (_Float16*)(wsb + 112 * MB); // 2MB each below
  _Float16* Wsak = (_Float16*)(wsb + 114 * MB);
  _Float16* Wsav = (_Float16*)(wsb + 116 * MB);
  _Float16* Wsao = (_Float16*)(wsb + 118 * MB);
  _Float16* Wcaq = (_Float16*)(wsb + 120 * MB);
  _Float16* Wcak = (_Float16*)(wsb + 122 * MB);
  _Float16* Wcav = (_Float16*)(wsb + 124 * MB);
  _Float16* Wcao = (_Float16*)(wsb + 126 * MB);
  _Float16* Wfc1 = (_Float16*)(wsb + 128 * MB); // 8MB
  _Float16* Wfc2 = (_Float16*)(wsb + 136 * MB); // 8MB
  float* OUT = (float*)d_out;

  dim3 blk(256);
  dim3 g_cv(NTOK * D_MODEL / 1024);
  dim3 g_t1(16, 16);                  // 1024x1024 weights
  dim3 g_tf1(64, 16);                 // fc1: K=1024, N=4096
  dim3 g_tf2(16, 64);                 // fc2: K=4096, N=1024
  dim3 g_1k(1024 / 128, NTOK / 128);  // GEMM, N=1024
  dim3 g_ff(D_FF / 128, NTOK / 128);  // GEMM, N=4096
  dim3 g_at(L_ / 128, B_ * N_HEADS);
  dim3 g_ln(NTOK);

  // ---- one-time f16 conversion / weight transpose ----
  cvt_f16_kernel<<<g_cv, blk, 0, stream>>>(x, xh, NTOK * D_MODEL);
  cvt_f16_kernel<<<g_cv, blk, 0, stream>>>(mem, memh, NTOK * D_MODEL);
  transpose_cvt_kernel<<<g_t1, blk, 0, stream>>>(sa_wq, Wsaq, 1024, 1024);
  transpose_cvt_kernel<<<g_t1, blk, 0, stream>>>(sa_wk, Wsak, 1024, 1024);
  transpose_cvt_kernel<<<g_t1, blk, 0, stream>>>(sa_wv, Wsav, 1024, 1024);
  transpose_cvt_kernel<<<g_t1, blk, 0, stream>>>(sa_wo, Wsao, 1024, 1024);
  transpose_cvt_kernel<<<g_t1, blk, 0, stream>>>(ca_wq, Wcaq, 1024, 1024);
  transpose_cvt_kernel<<<g_t1, blk, 0, stream>>>(ca_wk, Wcak, 1024, 1024);
  transpose_cvt_kernel<<<g_t1, blk, 0, stream>>>(ca_wv, Wcav, 1024, 1024);
  transpose_cvt_kernel<<<g_t1, blk, 0, stream>>>(ca_wo, Wcao, 1024, 1024);
  transpose_cvt_kernel<<<g_tf1, blk, 0, stream>>>(fc1_w, Wfc1, 1024, D_FF);
  transpose_cvt_kernel<<<g_tf2, blk, 0, stream>>>(fc2_w, Wfc2, D_FF, 1024);

  // ---- self attention ----
  gemm_f16_kernel<<<g_1k, blk, 0, stream>>>(xh, Wsaq, sa_bq, nullptr, Qh, NTOK, 1024, 1024, 0);
  gemm_f16_kernel<<<g_1k, blk, 0, stream>>>(xh, Wsak, sa_bk, nullptr, Kh, NTOK, 1024, 1024, 0);
  gemm_f16_kernel<<<g_1k, blk, 0, stream>>>(xh, Wsav, sa_bv, nullptr, Vh, NTOK, 1024, 1024, 0);
  attn_kernel<true><<<g_at, blk, 0, stream>>>(Qh, Kh, Vh, Ch, rel_emb, nullptr, L_, L_);
  gemm_f16_kernel<<<g_1k, blk, 0, stream>>>(Ch, Wsao, sa_bo, TMP, nullptr, NTOK, 1024, 1024, 0);
  add_ln_kernel<<<g_ln, blk, 0, stream>>>(x, TMP, ln1_g, ln1_b, X1f, X1h);

  // ---- cross attention ----
  gemm_f16_kernel<<<g_1k, blk, 0, stream>>>(X1h, Wcaq, ca_bq, nullptr, Qh, NTOK, 1024, 1024, 0);
  gemm_f16_kernel<<<g_1k, blk, 0, stream>>>(memh, Wcak, ca_bk, nullptr, Kh, NTOK, 1024, 1024, 0);
  gemm_f16_kernel<<<g_1k, blk, 0, stream>>>(memh, Wcav, ca_bv, nullptr, Vh, NTOK, 1024, 1024, 0);
  attn_kernel<false><<<g_at, blk, 0, stream>>>(Qh, Kh, Vh, Ch, nullptr, mem_mask, L_, L_);
  gemm_f16_kernel<<<g_1k, blk, 0, stream>>>(Ch, Wcao, ca_bo, TMP, nullptr, NTOK, 1024, 1024, 0);
  add_ln_kernel<<<g_ln, blk, 0, stream>>>(X1f, TMP, ln2_g, ln2_b, X2f, X2h);

  // ---- FFN ----
  gemm_f16_kernel<<<g_ff, blk, 0, stream>>>(X2h, Wfc1, fc1_b, nullptr, FFh, NTOK, D_FF, 1024, 1);
  gemm_f16_kernel<<<g_1k, blk, 0, stream>>>(FFh, Wfc2, fc2_b, TMP, nullptr, NTOK, 1024, D_FF, 0);
  add_ln_kernel<<<g_ln, blk, 0, stream>>>(X2f, TMP, ln3_g, ln3_b, OUT, nullptr);
}
